// ProductVectorQuantizer_14602888807210
// MI455X (gfx1250) — matrix operand order repaired and checked by reading, not compiled, not run on hardware
//
#include <hip/hip_runtime.h>
#include <hip/hip_bf16.h>
#include <stdint.h>

// Product VQ for MI455X (gfx1250, wave32).
// bf16 WMMA distance GEMM + TDM codebook staging + branchless argmin +
// fp32 gather/loss/encode. Memory-bound target: ~512MB / 23.3TB/s.

typedef __attribute__((ext_vector_type(16))) __bf16          v16bf;
typedef __attribute__((ext_vector_type(8)))  float           v8f;
typedef __attribute__((ext_vector_type(8)))  unsigned short  us8;
typedef __attribute__((ext_vector_type(4)))  unsigned int    u32x4;
typedef __attribute__((ext_vector_type(8)))  int             i32x8;
typedef __attribute__((ext_vector_type(4)))  int             i32x4;

#if __has_include(<hip/amd_detail/amd_gfx1250_TDM.h>)
#define TDM_6ARG 1
#else
#define TDM_6ARG 0
#endif

#define B_  64
#define T_  2048
#define D_  512
#define C_  4
#define K_  128
#define d_  128
#define N_  (B_ * T_)            // 131072 rows

#define ROWS_PER_WG    128
#define WAVES_PER_WG   8
#define ROWS_PER_WAVE  16

#define EBF_STRIDE 136           // ushorts per codeword row (128 + 8 pad = TDM pad 16B)
#define XS_STRIDE  132           // floats per x row (128 + 4 pad -> 4-bank shear)
#define EBF_BYTES  (K_ * EBF_STRIDE * 2)                            // 34816
#define XS_BYTES   (WAVES_PER_WG * ROWS_PER_WAVE * XS_STRIDE * 4)   // 67584
#define SMEM_BYTES (EBF_BYTES + XS_BYTES)                           // 102400 < 320KB/WGP

__device__ __forceinline__ unsigned short f2bf(float f) {
  union { float f; unsigned u; } x; x.f = f;
  unsigned u = x.u;
  unsigned r = (u + 0x7FFFu + ((u >> 16) & 1u)) >> 16;   // round-to-nearest-even
  return (unsigned short)r;
}

// Monotone map: float bits -> u32 preserving < order (incl. negatives).
__device__ __forceinline__ unsigned fkey(float f) {
  unsigned b = __float_as_uint(f);
  return b ^ (unsigned)(((int)b >> 31) | 0x80000000);
}

// ---- TDM: DMA one 128x128 bf16 codebook (row stride 128) into LDS with a
// 16B pad every 256B row -> LDS row stride 272B (= EBF_STRIDE ushorts). ----
__device__ __forceinline__ void tdm_load_codebook(const unsigned short* gsrc,
                                                  unsigned lds_off) {
  unsigned long long ga = (unsigned long long)(uintptr_t)gsrc;
  u32x4 g0;
  g0[0] = 1u;                                              // count=1 (user D#)
  g0[1] = lds_off;                                         // lds_addr
  g0[2] = (unsigned)(ga & 0xffffffffu);                    // global_addr[31:0]
  g0[3] = (unsigned)((ga >> 32) & 0x01ffffffu)             // global_addr[56:32]
        | (2u << 30);                                      // type = 2 (image)
  i32x8 g1;
  g1[0] = (1 << 16)        // data_size = 2 bytes (bf16)
        | (1 << 20)        // pad_enable
        | (5 << 22)        // pad_interval: 64 DWORDs = 256B
        | (3 << 25);       // pad_amount:   4 DWORDs = 16B
  g1[1] = (int)(128u << 16);  // tensor_dim0 = 128 (bits 79:48, low half)
  g1[2] = (int)(128u << 16);  // tensor_dim0 hi=0 | tensor_dim1 = 128 (low half)
  g1[3] = (int)(128u << 16);  // tensor_dim1 hi=0 | tile_dim0 = 128
  g1[4] = 128;                // tile_dim1 = 128 | tile_dim2 = 0
  g1[5] = 128;                // tensor_dim0_stride = 128 elements (contiguous)
  g1[6] = 0;
  g1[7] = 0;
  i32x4 z4 = {0, 0, 0, 0};
#if TDM_6ARG
  i32x8 z8 = {0, 0, 0, 0, 0, 0, 0, 0};
  __builtin_amdgcn_tensor_load_to_lds(g0, g1, z4, z4, z8, 0);
#else
  __builtin_amdgcn_tensor_load_to_lds(g0, g1, z4, z4, 0);
#endif
}

// ---- prep: bf16 codebook copy + e2 = sum(e*e) per codeword (into ws) ----
__global__ void pvq_prep(const float* __restrict__ emb,
                         unsigned short* __restrict__ ebf,
                         float* __restrict__ e2) {
  int t = blockIdx.x * blockDim.x + threadIdx.x;          // 0 .. C*K-1
  if (t >= C_ * K_) return;
  const float* src = emb + (size_t)t * d_;
  unsigned short* dst = ebf + (size_t)t * d_;
  float s = 0.f;
#pragma unroll 8
  for (int j = 0; j < d_; ++j) { float v = src[j]; s += v * v; dst[j] = f2bf(v); }
  e2[t] = s;
}

// ---- main fused kernel: 1024 blocks x 256 threads, 1 wave = 16 rows ----
__global__ void __launch_bounds__(256) pvq_main(
    const float* __restrict__ x,                 // N x D fp32
    const float* __restrict__ emb,               // C x K x d fp32 (L2-resident)
    const unsigned short* __restrict__ ebf_ws,   // C x K x d bf16
    const float* __restrict__ e2_ws,             // C x K fp32
    float* __restrict__ out_q,                   // N x D
    int*   __restrict__ out_enc,                 // N
    float* __restrict__ out_loss) {              // N
  extern __shared__ char smem[];
  unsigned short* ebf = (unsigned short*)smem;             // [K][EBF_STRIDE] bf16
  float* xs = (float*)(smem + EBF_BYTES);                  // [8][16][XS_STRIDE] f32

  const int tid  = threadIdx.x;
  const int wave = tid >> 5;
  const int lane = tid & 31;
  const int l15  = lane & 15;
  const int half = lane >> 4;
  const size_t rowbase = (size_t)blockIdx.x * ROWS_PER_WG + (size_t)wave * ROWS_PER_WAVE;
  const unsigned ebf_lds_off = (unsigned)(uintptr_t)(void*)ebf;  // LDS byte offset

  float* xw = xs + wave * (ROWS_PER_WAVE * XS_STRIDE);

  float lossacc = 0.f;
  int   enc = 0;

  for (int c = 0; c < C_; ++c) {
    __syncthreads();   // previous iteration done reading LDS

    // Kick the TDM for this c's bf16 codebook (one DMA per WG; EXEC-agnostic,
    // per-wave TENSORcnt -> wave 0 issues and waits, barrier publishes).
    if (wave == 0) {
      tdm_load_codebook(ebf_ws + (size_t)c * K_ * d_, ebf_lds_off);
    }

    // Meanwhile every wave stages its 16 x-rows (current sub-vector) in LDS.
#pragma unroll
    for (int m = 0; m < ROWS_PER_WAVE; ++m) {
      const float4 v = *(const float4*)(x + (rowbase + m) * D_ + c * d_ + lane * 4);
      *(float4*)(xw + m * XS_STRIDE + lane * 4) = v;
    }

    if (wave == 0) __builtin_amdgcn_s_wait_tensorcnt(0);
    __syncthreads();

    // Build A fragments (16x32 bf16) with native f32->bf16 converts (RTNE).
    // ISA layout: lane=M; lanes<16 hold K=kb+0..7 & kb+16..23, lanes>=16
    // hold K=kb+8..15 & kb+24..31.
    v16bf afr[4];
    {
      const int kh = half * 8;
      const float* xrow = xw + l15 * XS_STRIDE;
#pragma unroll
      for (int ks = 0; ks < 4; ++ks) {
        const int kb = ks * 32;
        v16bf av;
#pragma unroll
        for (int i = 0; i < 8; ++i) av[i]     = (__bf16)xrow[kb + kh + i];
#pragma unroll
        for (int i = 0; i < 8; ++i) av[8 + i] = (__bf16)xrow[kb + 16 + kh + i];
        afr[ks] = av;
      }
    }

    float bs[8];
    int   bk[8];
#pragma unroll
    for (int r = 0; r < 8; ++r) { bs[r] = 3.0e38f; bk[r] = 0; }

    const float* e2c = e2_ws + c * K_ + l15;

    // 8 column tiles of 16 codewords; d=128 -> 4 K-steps of 32. Fully
    // unrolled: e2 loads hoisted, no loop-carried SALU in the WMMA stream.
#pragma unroll
    for (int t = 0; t < 8; ++t) {
      v8f acc = {};
      // B fragment (32x16 bf16): lane=N (codeword 16t+l15); each lane half
      // holds K=kb+8*half+0..7 & kb+16+8*half+0..7 -> two ds_load_b128.
      const unsigned short* brow = ebf + (16 * t + l15) * EBF_STRIDE + half * 8;
#pragma unroll
      for (int ks = 0; ks < 4; ++ks) {
        union { v16bf v; us8 h[2]; } bf;
        bf.h[0] = *(const us8*)(brow + ks * 32);
        bf.h[1] = *(const us8*)(brow + ks * 32 + 16);
        acc = __builtin_amdgcn_wmma_f32_16x16x32_bf16(
            false, afr[ks], false, bf.v, (short)0, acc, false, false);
      }
      const float e2v = e2c[16 * t];
      const int   col = 16 * t + l15;
#pragma unroll
      for (int r = 0; r < 8; ++r) {
        float s = __builtin_fmaf(-2.0f, acc[r], e2v);  // x^2 row-constant: dropped
        if (s < bs[r]) { bs[r] = s; bk[r] = col; }     // strict < keeps smallest k
      }
    }

    // Branchless cross-lane argmin within each 16-lane half (C/D layout:
    // row r in lanes 0-15, row r+8 in lanes 16-31). Pack (score,idx) into a
    // sortable u64: ties resolve to smaller index, matching jnp.argmin.
#pragma unroll
    for (int r = 0; r < 8; ++r) {
      unsigned long long key =
          ((unsigned long long)fkey(bs[r]) << 32) | (unsigned)bk[r];
#pragma unroll
      for (int off = 1; off <= 8; off <<= 1) {
        unsigned long long o = __shfl_xor(key, off, 32);
        key = (o < key) ? o : key;
      }
      bk[r] = (int)(unsigned)key;
    }

    // Gather fp32 codeword (from L2), write quantized, exact fp32 loss, encode.
#pragma unroll
    for (int m = 0; m < ROWS_PER_WAVE; ++m) {
      const int r  = m & 7;
      const int km = __shfl(bk[r], (m >> 3) * 16, 32);
      const float4 q  = *(const float4*)(emb + ((size_t)(c * K_ + km)) * d_ + lane * 4);
      const float4 xv = *(const float4*)(xw + m * XS_STRIDE + lane * 4);
      *(float4*)(out_q + (rowbase + m) * D_ + c * d_ + lane * 4) = q;
      float d0 = q.x - xv.x, d1 = q.y - xv.y, d2 = q.z - xv.z, d3 = q.w - xv.w;
      float ss = d0 * d0 + d1 * d1 + d2 * d2 + d3 * d3;
#pragma unroll
      for (int off = 1; off < 32; off <<= 1) ss += __shfl_xor(ss, off, 32);
      if (lane == m) { lossacc += ss; enc = enc * K_ + km; }
    }
  }

  if (lane < 16) {
    const size_t n = rowbase + lane;
    out_loss[n] = 1.25f * lossacc;   // (1 + COMMITMENT_COST) * ||q - x||^2
    out_enc[n]  = enc;               // radix-128 packed, c=0 most significant
  }
}

extern "C" void kernel_launch(void* const* d_in, const int* in_sizes, int n_in,
                              void* d_out, int out_size, void* d_ws, size_t ws_size,
                              hipStream_t stream) {
  (void)in_sizes; (void)n_in; (void)out_size; (void)ws_size;
  const float* x   = (const float*)d_in[0];     // (B,T,D) f32
  const float* emb = (const float*)d_in[1];     // (C,K,d) f32

  unsigned short* ebf_ws = (unsigned short*)d_ws;                       // C*K*d bf16
  float* e2_ws = (float*)((char*)d_ws + (size_t)C_ * K_ * d_ * sizeof(unsigned short));

  // d_out = [quantized_sg: N*D f32][encoding_indices: N i32][quantization_loss: N f32]
  float* out_q    = (float*)d_out;
  int*   out_enc  = (int*)(out_q + (size_t)N_ * D_);
  float* out_loss = (float*)(out_enc + N_);

  pvq_prep<<<(C_ * K_ + 255) / 256, 256, 0, stream>>>(emb, ebf_ws, e2_ws);
  pvq_main<<<N_ / ROWS_PER_WG, 256, SMEM_BYTES, stream>>>(
      x, emb, ebf_ws, e2_ws, out_q, out_enc, out_loss);
}
